// PointerNetwork_63488206569933
// MI455X (gfx1250) — matrix operand order repaired and checked by reading, not compiled, not run on hardware
//
#include <hip/hip_runtime.h>
#include <hip/hip_bf16.h>

// ---------------------------------------------------------------------------
// Pointer network on MI455X (gfx1250, wave32).
//   B=1024, S=256, H=256, U=10.
// bf16 WMMA (v_wmma_f32_16x16x32_bf16) for all LSTM gate GEMMs, f32 cell
// state + nonlinearities; enc_out stored bf16 to halve the dominant
// attention-context HBM traffic; one kernel launch per time step
// (graph-captured), ping-pong h buffers by step parity.
// Each wave runs TWO independent WMMA accumulator chains (two 16x16 column
// tiles sharing one A fragment) to avoid the bf16 WMMA->WMMA RAW stall.
// ---------------------------------------------------------------------------

#define Bsz 1024
#define Sd  256
#define Hd  256
#define Ud  10

typedef __attribute__((ext_vector_type(16))) __bf16 v16bf;
typedef __attribute__((ext_vector_type(8)))  __bf16 v8bf;
typedef __attribute__((ext_vector_type(8)))  float  v8f;

__device__ __forceinline__ float sigm(float x) { return 1.0f / (1.0f + expf(-x)); }

// ---------------------------------------------------------------------------
// Pack a weight matrix W[4H x ldw] (row-major f32) into bf16 B-fragments for
// v_wmma_f32_16x16x32_bf16.  B-matrix is K x N (K = contraction over H,
// N = gate-output column).  This file's own k order (A uses the same): lane = (n, khalf),
// element e holds K = khalf*16 + e.  Each (kblock, jtile) fragment is stored
// as 32 lanes x 16 bf16 contiguous, so the GEMM loop does one v16bf load.
// ---------------------------------------------------------------------------
__global__ __launch_bounds__(256)
void pack_b_kernel(const float* __restrict__ W, int ldw, __bf16* __restrict__ out)
{
    int idx  = blockIdx.x * 256 + threadIdx.x;      // 8 kb * 64 jb * 32 lanes * 16 e = 262144
    int e    = idx & 15;
    int lane = (idx >> 4) & 31;
    int jb   = (idx >> 9) & 63;
    int kb   = idx >> 15;
    int n    = lane & 15;
    int kh   = lane >> 4;
    int j    = jb * 16 + n;               // column in 4H gate space
    int k    = kb * 32 + kh * 16 + e;     // contraction index (< 256)
    out[idx] = (__bf16)W[(size_t)j * ldw + k];
}

// ---------------------------------------------------------------------------
// Fused LSTM step:  G = Hbf @ Bpk_hh  (+ A2bf @ Bpk_ih)  + bias + r1x*r1w,
// then i/f/g/o nonlinearities, cell update, write new c (f32, in-place) and
// new h (bf16, ping-pong), optionally streaming h into enc_out (bf16).
//
// Grid: (H/32, B/32), block 256 = 8 waves.  Wave w: gate = w&3, row-half =
// w>>2; each wave owns TWO 16x16 tiles (adjacent u-columns) -> two
// independent accumulator chains, one shared A fragment per K-chunk.
// A-fragment layout (16-bit A 16x32): lane=(m, khalf); elements 0..7 cover
// K = khalf*16+0..7, elements 8..15 cover K = khalf*16+8..15 -> two
// contiguous 16-byte loads per lane per K-chunk.
// ---------------------------------------------------------------------------
__global__ __launch_bounds__(256)
void lstm_step_wmma(const __bf16* __restrict__ Abf,      // [B,H] h_prev (bf16)
                    const __bf16* __restrict__ Bpk_hh,   // packed Whh^T
                    const __bf16* __restrict__ A2bf,     // [B,H] di (decoder) or null
                    const __bf16* __restrict__ Bpk_ih,   // packed Wih^T (decoder)
                    const float*  __restrict__ bias,     // [4H]
                    const float*  __restrict__ r1w, int r1w_stride,  // rank-1 weight col
                    const float*  __restrict__ r1x, int r1x_stride,  // rank-1 per-row scalar
                    float*        __restrict__ c_state,  // [B,H] in/out
                    __bf16*       __restrict__ hbf_out,  // [B,H] out
                    __bf16*       __restrict__ enc_slice,// &enc_out[t*H] or null
                    int second_gemm)
{
    __shared__ float lds_g[4][32][32];                   // 16 KB

    const int tid   = threadIdx.x;
    const int lane  = tid & 31;
    const int wid   = tid >> 5;
    const int gate  = wid & 3;
    const int mhalf = wid >> 2;
    const int bx    = blockIdx.x;              // 32-wide u-tile: 0..7
    const int b0    = blockIdx.y * 32;         // batch tile base
    const int jb0   = gate * 16 + bx * 2;      // column tiles in 4H space
    const int jb1   = jb0 + 1;

    const int m  = lane & 15;
    const int kh = lane >> 4;
    const int arow = b0 + mhalf * 16 + m;

    v8f acc0 = {};
    v8f acc1 = {};

    #pragma unroll
    for (int kb = 0; kb < Hd / 32; ++kb) {
        const __bf16* ap = Abf + (size_t)arow * Hd + kb * 32 + kh * 16;
        v8bf lo = *(const v8bf*)ap;
        v8bf hi = *(const v8bf*)(ap + 8);
        v16bf afrag;
        #pragma unroll
        for (int e = 0; e < 8; ++e) { afrag[e] = lo[e]; afrag[8 + e] = hi[e]; }
        v16bf bf0 = *(const v16bf*)(Bpk_hh + (((size_t)kb * 64 + jb0) * 32 + lane) * 16);
        v16bf bf1 = *(const v16bf*)(Bpk_hh + (((size_t)kb * 64 + jb1) * 32 + lane) * 16);
        acc0 = __builtin_amdgcn_wmma_f32_16x16x32_bf16(false, afrag, false, bf0,
                                                       (short)0, acc0, false, false);
        acc1 = __builtin_amdgcn_wmma_f32_16x16x32_bf16(false, afrag, false, bf1,
                                                       (short)0, acc1, false, false);
    }
    if (second_gemm) {  // decoder: + di @ dec_Wih[:, :H]^T   (uniform branch)
        #pragma unroll
        for (int kb = 0; kb < Hd / 32; ++kb) {
            const __bf16* ap = A2bf + (size_t)arow * Hd + kb * 32 + kh * 16;
            v8bf lo = *(const v8bf*)ap;
            v8bf hi = *(const v8bf*)(ap + 8);
            v16bf afrag;
            #pragma unroll
            for (int e = 0; e < 8; ++e) { afrag[e] = lo[e]; afrag[8 + e] = hi[e]; }
            v16bf bf0 = *(const v16bf*)(Bpk_ih + (((size_t)kb * 64 + jb0) * 32 + lane) * 16);
            v16bf bf1 = *(const v16bf*)(Bpk_ih + (((size_t)kb * 64 + jb1) * 32 + lane) * 16);
            acc0 = __builtin_amdgcn_wmma_f32_16x16x32_bf16(false, afrag, false, bf0,
                                                           (short)0, acc0, false, false);
            acc1 = __builtin_amdgcn_wmma_f32_16x16x32_bf16(false, afrag, false, bf1,
                                                           (short)0, acc1, false, false);
        }
    }

    // C/D layout: lanes 0-15 -> N=lane, M=r; lanes 16-31 -> N=lane-16, M=8+r.
    const int n    = lane & 15;
    const int mtop = lane >> 4;
    const int j0   = gate * Hd + bx * 32 + n;        // tile 0 column in 4H
    const int j1   = j0 + 16;                        // tile 1 column
    const float bj0 = bias[j0];
    const float bj1 = bias[j1];
    const float wj0 = r1w[(size_t)j0 * r1w_stride];
    const float wj1 = r1w[(size_t)j1 * r1w_stride];

    #pragma unroll
    for (int r = 0; r < 8; ++r) {
        int mrow = mhalf * 16 + mtop * 8 + r;        // 0..31 within block
        int b    = b0 + mrow;
        float xs = r1x[(size_t)b * r1x_stride];
        lds_g[gate][mrow][n]      = acc0[r] + bj0 + xs * wj0;
        lds_g[gate][mrow][n + 16] = acc1[r] + bj1 + xs * wj1;
    }
    __syncthreads();

    #pragma unroll
    for (int it = 0; it < 4; ++it) {
        int idx  = tid + it * 256;                   // 1024 cells: 32 rows x 32 cols
        int mrow = idx >> 5, nn = idx & 31;
        int b = b0 + mrow, u = bx * 32 + nn;
        float iv = lds_g[0][mrow][nn];
        float fv = lds_g[1][mrow][nn];
        float gv = lds_g[2][mrow][nn];
        float ov = lds_g[3][mrow][nn];
        size_t ci = (size_t)b * Hd + u;
        float cn = sigm(fv) * c_state[ci] + sigm(iv) * tanhf(gv);
        float hn = sigm(ov) * tanhf(cn);
        c_state[ci] = cn;
        hbf_out[ci] = (__bf16)hn;
        if (enc_slice) enc_slice[(size_t)b * Sd * Hd + u] = (__bf16)hn;
    }
}

// ---------------------------------------------------------------------------
// enc_proj[b,s,u] = sum_h enc_bf[b,s,h] * W1[u,h]    (once; W1 cached in LDS)
// ---------------------------------------------------------------------------
__global__ __launch_bounds__(256)
void attn_proj_kernel(const __bf16* __restrict__ enc, const float* __restrict__ W1,
                      float* __restrict__ proj)
{
    __shared__ float w1s[Ud][Hd];
    for (int i = threadIdx.x; i < Ud * Hd; i += 256) w1s[i / Hd][i % Hd] = W1[i];
    __syncthreads();

    int row = blockIdx.x * 256 + threadIdx.x;          // (b*S + s), 0..B*S-1
    const __bf16* er = enc + (size_t)row * Hd;
    float accu[Ud] = {};
    for (int h8 = 0; h8 < Hd; h8 += 8) {
        v8bf ev = *(const v8bf*)(er + h8);
        #pragma unroll
        for (int e = 0; e < 8; ++e) {
            float f = (float)ev[e];
            #pragma unroll
            for (int u = 0; u < Ud; ++u) accu[u] += f * w1s[u][h8 + e];
        }
    }
    float* pr = proj + (size_t)row * Ud;
    #pragma unroll
    for (int u = 0; u < Ud; ++u) pr[u] = accu[u];
}

// ---------------------------------------------------------------------------
// q[b,u] = sum_h h_dec[b,h] * W2[u,h]    (tiny; one block per batch row)
// ---------------------------------------------------------------------------
__global__ __launch_bounds__(256)
void dec_q_kernel(const __bf16* __restrict__ hbf, const float* __restrict__ W2,
                  float* __restrict__ q)
{
    int b = blockIdx.x, tid = threadIdx.x;
    float hv = (float)hbf[(size_t)b * Hd + tid];
    __shared__ float red[8][Ud];
    #pragma unroll
    for (int u = 0; u < Ud; ++u) {
        float v = hv * W2[u * Hd + tid];
        for (int o = 16; o >= 1; o >>= 1) v += __shfl_xor(v, o, 32);
        if ((tid & 31) == 0) red[tid >> 5][u] = v;
    }
    __syncthreads();
    if (tid < Ud) {
        float s = 0.f;
        #pragma unroll
        for (int w = 0; w < 8; ++w) s += red[w][tid];
        q[(size_t)b * 16 + tid] = s;
    }
}

// ---------------------------------------------------------------------------
// logits[b,s] = sum_u v[u] * tanh(enc_proj[b,s,u] + q[b,u])
// ---------------------------------------------------------------------------
__global__ __launch_bounds__(256)
void attn_logits_kernel(const float* __restrict__ proj, const float* __restrict__ q,
                        const float* __restrict__ V, float* __restrict__ logits)
{
    int b = blockIdx.x, s = threadIdx.x;
    __shared__ float qs[Ud], vs[Ud];
    if (threadIdx.x < Ud) { qs[threadIdx.x] = q[(size_t)b * 16 + threadIdx.x];
                            vs[threadIdx.x] = V[threadIdx.x]; }
    __syncthreads();
    const float* pr = proj + ((size_t)b * Sd + s) * Ud;
    float acc = 0.f;
    #pragma unroll
    for (int u = 0; u < Ud; ++u) acc += vs[u] * tanhf(pr[u] + qs[u]);
    logits[(size_t)b * Sd + s] = acc;
}

// ---------------------------------------------------------------------------
// softmax + argmax + CE + teacher-forced next decoder input.  One block per b.
// ---------------------------------------------------------------------------
__global__ __launch_bounds__(256)
void softmax_kernel(const float* __restrict__ logits, const int* __restrict__ y,
                    const float* __restrict__ x, int step,
                    float* __restrict__ attw, float* __restrict__ dec_in_next,
                    float* __restrict__ preds_out, float* __restrict__ loss_out)
{
    int b = blockIdx.x, s = threadIdx.x;
    float l = logits[(size_t)b * Sd + s];
    __shared__ float smax[256]; __shared__ int sidx[256]; __shared__ float ssum[256];
    smax[s] = l; sidx[s] = s;
    __syncthreads();
    for (int o = 128; o >= 1; o >>= 1) {
        if (s < o) {
            float a = smax[s], bv = smax[s + o];
            if (bv > a || (bv == a && sidx[s + o] < sidx[s])) { smax[s] = bv; sidx[s] = sidx[s + o]; }
        }
        __syncthreads();
    }
    float mx = smax[0];
    float e = expf(l - mx);
    ssum[s] = e;
    __syncthreads();
    for (int o = 128; o >= 1; o >>= 1) { if (s < o) ssum[s] += ssum[s + o]; __syncthreads(); }
    float sum = ssum[0];
    attw[(size_t)b * Sd + s] = e / sum;
    if (s == 0) {
        int label = y[(size_t)b * Sd + step];
        float lp = logits[(size_t)b * Sd + label] - mx - logf(sum);
        atomicAdd(loss_out, -lp * (1.0f / ((float)Bsz * (float)Bsz)));
        preds_out[(size_t)step * Bsz + b] = (float)sidx[0];
        dec_in_next[b] = x[(size_t)b * Sd + label];
    }
}

// ---------------------------------------------------------------------------
// di[b,h] = sum_s a[b,s] * enc_bf[b,s,h]   (the HBM-dominant reduction).
// Coalesced bf16 rows; prefetch 8 rows ahead (global_prefetch_b8).
// ---------------------------------------------------------------------------
__global__ __launch_bounds__(256)
void attn_di_kernel(const float* __restrict__ attw, const __bf16* __restrict__ enc,
                    __bf16* __restrict__ dibf)
{
    int b = blockIdx.x, h = threadIdx.x;
    __shared__ float aw[Sd];
    aw[h] = attw[(size_t)b * Sd + h];
    __syncthreads();
    const __bf16* eb = enc + (size_t)b * Sd * Hd + h;
    float acc = 0.f;
    for (int s = 0; s < Sd; ++s) {
        if (s + 8 < Sd) __builtin_prefetch(eb + (size_t)(s + 8) * Hd, 0, 1);
        acc += aw[s] * (float)eb[(size_t)s * Hd];
    }
    dibf[(size_t)b * Hd + h] = (__bf16)acc;
}

__global__ void zero_kernel(unsigned int* p, unsigned int nwords)
{
    unsigned int i = blockIdx.x * 256u + threadIdx.x;
    if (i < nwords) p[i] = 0u;
}

// ---------------------------------------------------------------------------
extern "C" void kernel_launch(void* const* d_in, const int* in_sizes, int n_in,
                              void* d_out, int out_size, void* d_ws, size_t ws_size,
                              hipStream_t stream)
{
    (void)in_sizes; (void)n_in; (void)out_size; (void)ws_size;
    const float* x       = (const float*)d_in[0];
    const int*   y       = (const int*)  d_in[1];
    const float* enc_Wih = (const float*)d_in[2];
    const float* enc_Whh = (const float*)d_in[3];
    const float* enc_b   = (const float*)d_in[4];
    const float* dec_Wih = (const float*)d_in[5];
    const float* dec_Whh = (const float*)d_in[6];
    const float* dec_b   = (const float*)d_in[7];
    const float* W1      = (const float*)d_in[8];
    const float* W2      = (const float*)d_in[9];
    const float* V       = (const float*)d_in[10];
    float* out = (float*)d_out;                  // [S*B preds] then [1 loss]

    // ---- workspace carving (256B aligned) ----
    char* base = (char*)d_ws;
    size_t off = 0;
    auto carve = [&](size_t bytes) -> char* {
        char* p = base + off; off = (off + bytes + 255) & ~(size_t)255; return p;
    };
    __bf16* enc_out_bf = (__bf16*)carve((size_t)Bsz * Sd * Hd * 2);   // 128 MB
    __bf16* Bpk_enc    = (__bf16*)carve((size_t)4 * Hd * Hd * 2);
    __bf16* Bpk_dec_hh = (__bf16*)carve((size_t)4 * Hd * Hd * 2);
    __bf16* Bpk_dec_ih = (__bf16*)carve((size_t)4 * Hd * Hd * 2);
    float*  enc_proj   = (float*) carve((size_t)Bsz * Sd * Ud * 4);   // 10.5 MB
    __bf16* hbf_enc0   = (__bf16*)carve((size_t)Bsz * Hd * 2);
    __bf16* hbf_enc1   = (__bf16*)carve((size_t)Bsz * Hd * 2);
    float*  c_enc      = (float*) carve((size_t)Bsz * Hd * 4);
    __bf16* hbf_dec0   = (__bf16*)carve((size_t)Bsz * Hd * 2);
    __bf16* hbf_dec1   = (__bf16*)carve((size_t)Bsz * Hd * 2);
    float*  c_dec      = (float*) carve((size_t)Bsz * Hd * 4);
    __bf16* dibf       = (__bf16*)carve((size_t)Bsz * Hd * 2);
    float*  q          = (float*) carve((size_t)Bsz * 16 * 4);
    float*  logits     = (float*) carve((size_t)Bsz * Sd * 4);
    float*  attw       = (float*) carve((size_t)Bsz * Sd * 4);
    float*  dec_in0    = (float*) carve((size_t)Bsz * 4);
    float*  dec_in1    = (float*) carve((size_t)Bsz * 4);
    __bf16* hbf_enc[2] = { hbf_enc0, hbf_enc1 };
    __bf16* hbf_dec[2] = { hbf_dec0, hbf_dec1 };
    float*  dec_in[2]  = { dec_in0, dec_in1 };

    // ---- init: zero states + loss slot (d_out/d_ws are poisoned) ----
    auto zero = [&](void* p, size_t bytes) {
        unsigned int nw = (unsigned int)(bytes / 4);
        zero_kernel<<<(nw + 255u) / 256u, 256, 0, stream>>>((unsigned int*)p, nw);
    };
    zero(c_enc,      (size_t)Bsz * Hd * 4);
    zero(c_dec,      (size_t)Bsz * Hd * 4);
    zero(hbf_enc0,   (size_t)Bsz * Hd * 2);
    zero(hbf_dec0,   (size_t)Bsz * Hd * 2);
    zero(dec_in0,    (size_t)Bsz * 4);
    zero(out + (size_t)Sd * Bsz, 4);

    // ---- pack weights into WMMA B-fragment layout (bf16) ----
    pack_b_kernel<<<1024, 256, 0, stream>>>(enc_Whh, Hd,     Bpk_enc);
    pack_b_kernel<<<1024, 256, 0, stream>>>(dec_Whh, Hd,     Bpk_dec_hh);
    pack_b_kernel<<<1024, 256, 0, stream>>>(dec_Wih, Hd + 1, Bpk_dec_ih);

    dim3 lstm_grid(Hd / 32, Bsz / 32);

    // ---- encoder: 256 sequential WMMA LSTM steps ----
    for (int t = 0; t < Sd; ++t) {
        lstm_step_wmma<<<lstm_grid, 256, 0, stream>>>(
            hbf_enc[t & 1], Bpk_enc, nullptr, nullptr,
            enc_b, enc_Wih, 1, x + t, Sd,
            c_enc, hbf_enc[(t + 1) & 1],
            enc_out_bf + (size_t)t * Hd, 0);
    }

    // ---- attention projection (once) ----
    attn_proj_kernel<<<(Bsz * Sd) / 256, 256, 0, stream>>>(enc_out_bf, W1, enc_proj);

    // ---- decoder: 256 sequential steps ----
    for (int s = 0; s < Sd; ++s) {
        dec_q_kernel      <<<Bsz, 256, 0, stream>>>(hbf_dec[s & 1], W2, q);
        attn_logits_kernel<<<Bsz, 256, 0, stream>>>(enc_proj, q, V, logits);
        softmax_kernel    <<<Bsz, 256, 0, stream>>>(logits, y, x, s, attw,
                                                    dec_in[(s + 1) & 1], out,
                                                    out + (size_t)Sd * Bsz);
        attn_di_kernel    <<<Bsz, 256, 0, stream>>>(attw, enc_out_bf, dibf);
        lstm_step_wmma    <<<lstm_grid, 256, 0, stream>>>(
            hbf_dec[s & 1], Bpk_dec_hh, dibf, Bpk_dec_ih,
            dec_b, dec_Wih + Hd, Hd + 1, dec_in[s & 1], 1,
            c_dec, hbf_dec[(s + 1) & 1], nullptr, 1);
    }
}
